// GCNModel_42528766165363
// MI455X (gfx1250) — matrix operand order repaired and checked by reading, not compiled, not run on hardware
//
#include <hip/hip_runtime.h>
#include <hip/hip_bf16.h>

#define N_NODES 100000
#define N_EDGES 800000
#define D_INF   128
#define H_DIM   128
#define M_DIM   256
#define L_DIM   40
#define L_PAD   48

typedef __attribute__((ext_vector_type(16))) __bf16 v16bf;
typedef __attribute__((ext_vector_type(8)))  __bf16 v8bf;
typedef __attribute__((ext_vector_type(8)))  float  v8f;

// ---------------- graph-norm kernels ----------------

__global__ void k_deg_init(float* deg) {
    int i = blockIdx.x * blockDim.x + threadIdx.x;
    if (i < N_NODES) deg[i] = 1.0f;          // self-loop weight
}

__global__ void k_deg_accum(const int* __restrict__ row,
                            const float* __restrict__ w,
                            float* __restrict__ deg) {
    int e = blockIdx.x * blockDim.x + threadIdx.x;
    if (e < N_EDGES) atomicAdd(&deg[row[e]], w[e]);
}

__global__ void k_dinv(float* deg) {
    int i = blockIdx.x * blockDim.x + threadIdx.x;
    if (i < N_NODES) {
        float d = deg[i];
        deg[i] = (d > 0.0f) ? rsqrtf(d) : 0.0f;
    }
}

__global__ void k_norm(const int* __restrict__ row, const int* __restrict__ col,
                       const float* __restrict__ w, const float* __restrict__ dinv,
                       float* __restrict__ norm) {
    int e = blockIdx.x * blockDim.x + threadIdx.x;
    if (e < N_EDGES) norm[e] = dinv[row[e]] * w[e] * dinv[col[e]];
}

// ---------------- conversion kernels ----------------

__global__ void k_f2bf(const float* __restrict__ src, __bf16* __restrict__ dst, int n) {
    int i = blockIdx.x * blockDim.x + threadIdx.x;
    if (i < n) dst[i] = (__bf16)src[i];
}

// W is [K, Nc] row-major fp32 -> WT is [NcPad, K] row-major bf16 (transposed, zero-padded cols)
__global__ void k_wtrans(const float* __restrict__ W, __bf16* __restrict__ WT,
                         int K, int Nc, int NcPad) {
    int idx = blockIdx.x * blockDim.x + threadIdx.x;
    int total = NcPad * K;
    if (idx < total) {
        int n = idx / K;
        int k = idx - n * K;
        WT[idx] = (n < Nc) ? (__bf16)W[(size_t)k * Nc + n] : (__bf16)0.0f;
    }
}

// ---------------- aggregation kernels ----------------

// agg[i, :] = xw[i, :] * dinv[i]^2  (the self-loop message; also serves as init)
__global__ void k_agg_init(const float* __restrict__ xw, const float* __restrict__ dinv,
                           float* __restrict__ agg) {
    int idx = blockIdx.x * blockDim.x + threadIdx.x;
    if (idx < N_NODES * H_DIM) {
        int i = idx >> 7;
        float di = dinv[i];
        agg[idx] = xw[idx] * di * di;
    }
}

// one lane per (edge, 4-feature chunk): gather float4 from col, scatter atomics to row
__global__ void k_agg_edges(const int* __restrict__ row, const int* __restrict__ col,
                            const float* __restrict__ norm, const float* __restrict__ xw,
                            float* __restrict__ agg) {
    int gid = blockIdx.x * blockDim.x + threadIdx.x;
    int e = gid >> 5;
    int q = gid & 31;
    if (e < N_EDGES) {
        float nv = norm[e];
        const float4 v = *(const float4*)(xw + (size_t)col[e] * H_DIM + q * 4);
        float* dst = agg + (size_t)row[e] * H_DIM + q * 4;
        atomicAdd(dst + 0, v.x * nv);
        atomicAdd(dst + 1, v.y * nv);
        atomicAdd(dst + 2, v.z * nv);
        atomicAdd(dst + 3, v.w * nv);
    }
}

// h = relu(agg + b) -> bf16 (next layer's GEMM input)
__global__ void k_bias_relu_bf(const float* __restrict__ agg, const float* __restrict__ b,
                               __bf16* __restrict__ hbf) {
    int idx = blockIdx.x * blockDim.x + threadIdx.x;
    if (idx < N_NODES * H_DIM) {
        int f = idx & (H_DIM - 1);
        float v = agg[idx] + b[f];
        hbf[idx] = (__bf16)fmaxf(v, 0.0f);
    }
}

// ---------------- WMMA GEMM ----------------
// C[rows, Nc] = A[rows, K] (bf16, row-major) x B[K, Nc]  with B given pre-transposed
// as BT[Nc(Pad), K] bf16 row-major. One wave computes a 16 x (16*NT) tile.
// mode 0: store fp32 Cf                    (GCN layer pre-aggregation)
// mode 1: v += bias; relu; store bf16 Cb   (MLP hidden)
// mode 2: v += bias; store fp32 Cf[row*outW+col] for col < outW (final logits)
template <int NT>
__global__ __launch_bounds__(256) void k_gemm_bf16(
    const __bf16* __restrict__ A, const __bf16* __restrict__ BT,
    const float* __restrict__ bias,
    float* __restrict__ Cf, __bf16* __restrict__ Cb,
    int rows, int K, int Nc, int mode, int outW) {

    const int rowTiles  = rows >> 4;
    const int colSupers = Nc / (16 * NT);
    const int wave = blockIdx.x * (blockDim.x >> 5) + (threadIdx.x >> 5);
    if (wave >= rowTiles * colSupers) return;          // wave-uniform: EXEC stays full

    const int rt = wave % rowTiles;
    const int cs = wave / rowTiles;
    const int row0 = rt << 4;
    const int col0 = cs * 16 * NT;

    const int lane = threadIdx.x & 31;
    const int hh   = lane >> 4;      // lane half (0/1)
    const int mn   = lane & 15;      // M index for A, N index for B/C

    v8f acc[NT];
#pragma unroll
    for (int t = 0; t < NT; ++t)
#pragma unroll
        for (int j = 0; j < 8; ++j) acc[t][j] = 0.0f;

    const __bf16* arow = A + (size_t)(row0 + mn) * K;

    for (int kb = 0; kb < K; kb += 32) {
        // A fragment: elems j -> K = kb + hh*16 + j (matches B)
        v16bf a = *(const v16bf*)(arow + kb + hh * 16);

#pragma unroll
        for (int t = 0; t < NT; ++t) {
            // B fragment: elems j -> K = kb + hh*16 + j, column = col0 + t*16 + mn
            const __bf16* bp = BT + (size_t)(col0 + t * 16 + mn) * K + kb + hh * 16;
            v16bf b = *(const v16bf*)bp;
            acc[t] = __builtin_amdgcn_wmma_f32_16x16x32_bf16(
                false, a, false, b, (short)0, acc[t], false, false);
        }
    }

#pragma unroll
    for (int t = 0; t < NT; ++t) {
        const int col = col0 + t * 16 + mn;
        float bv = 0.0f;
        if (mode != 0) bv = (mode == 2 && col >= outW) ? 0.0f : bias[col];
#pragma unroll
        for (int r = 0; r < 8; ++r) {
            const int row = row0 + hh * 8 + r;   // D layout: VGPR r -> M = r + hh*8
            float v = acc[t][r] + bv;
            if (mode == 1) v = fmaxf(v, 0.0f);
            if (mode == 2) {
                if (col < outW) Cf[(size_t)row * outW + col] = v;
            } else {
                if (Cf) Cf[(size_t)row * Nc + col] = v;
                if (Cb) Cb[(size_t)row * Nc + col] = (__bf16)v;
            }
        }
    }
}

// ---------------- launcher ----------------

extern "C" void kernel_launch(void* const* d_in, const int* in_sizes, int n_in,
                              void* d_out, int out_size, void* d_ws, size_t ws_size,
                              hipStream_t stream) {
    const float* x   = (const float*)d_in[0];
    const int*   ei  = (const int*)  d_in[1];
    const float* ew  = (const float*)d_in[2];
    const float* W0  = (const float*)d_in[3];
    const float* b0  = (const float*)d_in[4];
    const float* W1  = (const float*)d_in[5];
    const float* b1  = (const float*)d_in[6];
    const float* W2  = (const float*)d_in[7];
    const float* b2  = (const float*)d_in[8];
    const float* Wm0 = (const float*)d_in[9];
    const float* bm0 = (const float*)d_in[10];
    const float* Wm1 = (const float*)d_in[11];
    const float* bm1 = (const float*)d_in[12];

    const int* rowI = ei;             // edge_index[0]
    const int* colI = ei + N_EDGES;   // edge_index[1]

    char* p = (char*)d_ws;
    auto take = [&](size_t bytes) -> char* {
        char* r = p;
        p += (bytes + 255) & ~(size_t)255;
        return r;
    };
    float*  dinv = (float*) take((size_t)N_NODES * 4);
    float*  norm = (float*) take((size_t)N_EDGES * 4);
    __bf16* hbf  = (__bf16*)take((size_t)N_NODES * H_DIM * 2);
    float*  xw   = (float*) take((size_t)N_NODES * H_DIM * 4);
    float*  agg  = (float*) take((size_t)N_NODES * H_DIM * 4);
    __bf16* mbf  = (__bf16*)take((size_t)N_NODES * M_DIM * 2);
    __bf16* W0T  = (__bf16*)take((size_t)H_DIM * D_INF * 2);
    __bf16* W1T  = (__bf16*)take((size_t)H_DIM * H_DIM * 2);
    __bf16* W2T  = (__bf16*)take((size_t)H_DIM * H_DIM * 2);
    __bf16* Wm0T = (__bf16*)take((size_t)M_DIM * H_DIM * 2);
    __bf16* Wm1T = (__bf16*)take((size_t)L_PAD * M_DIM * 2);

    const int BS = 256;
    auto blocks = [](long long n, int bs) { return (unsigned)((n + bs - 1) / bs); };

    // --- graph norm ---
    k_deg_init <<<blocks(N_NODES, BS), BS, 0, stream>>>(dinv);
    k_deg_accum<<<blocks(N_EDGES, BS), BS, 0, stream>>>(rowI, ew, dinv);
    k_dinv     <<<blocks(N_NODES, BS), BS, 0, stream>>>(dinv);
    k_norm     <<<blocks(N_EDGES, BS), BS, 0, stream>>>(rowI, colI, ew, dinv, norm);

    // --- one-time conversions ---
    k_f2bf  <<<blocks((long long)N_NODES * D_INF, BS), BS, 0, stream>>>(x, hbf, N_NODES * D_INF);
    k_wtrans<<<blocks(H_DIM * D_INF, BS), BS, 0, stream>>>(W0,  W0T,  D_INF, H_DIM, H_DIM);
    k_wtrans<<<blocks(H_DIM * H_DIM, BS), BS, 0, stream>>>(W1,  W1T,  H_DIM, H_DIM, H_DIM);
    k_wtrans<<<blocks(H_DIM * H_DIM, BS), BS, 0, stream>>>(W2,  W2T,  H_DIM, H_DIM, H_DIM);
    k_wtrans<<<blocks(M_DIM * H_DIM, BS), BS, 0, stream>>>(Wm0, Wm0T, H_DIM, M_DIM, M_DIM);
    k_wtrans<<<blocks(L_PAD * M_DIM, BS), BS, 0, stream>>>(Wm1, Wm1T, M_DIM, L_DIM, L_PAD);

    // --- 3 GCN layers ---
    const __bf16* WT[3]   = {W0T, W1T, W2T};
    const float*  bias[3] = {b0, b1, b2};
    for (int l = 0; l < 3; ++l) {
        // xw = h @ W  (WMMA)
        {
            int waves = (N_NODES >> 4) * (H_DIM / 64);   // NT=4 -> 16x64 per wave
            k_gemm_bf16<4><<<blocks(waves * 32LL, BS), BS, 0, stream>>>(
                hbf, WT[l], nullptr, xw, nullptr, N_NODES, H_DIM, H_DIM, 0, 0);
        }
        // agg = self-loop init + edge scatter
        k_agg_init <<<blocks((long long)N_NODES * H_DIM, BS), BS, 0, stream>>>(xw, dinv, agg);
        k_agg_edges<<<blocks((long long)N_EDGES * 32, BS), BS, 0, stream>>>(rowI, colI, norm, xw, agg);
        // h = relu(agg + b) -> bf16
        k_bias_relu_bf<<<blocks((long long)N_NODES * H_DIM, BS), BS, 0, stream>>>(agg, bias[l], hbf);
    }

    // --- MLP hidden: mbf = relu(h @ Wm0 + bm0) in bf16 ---
    {
        int waves = (N_NODES >> 4) * (M_DIM / 64);       // NT=4
        k_gemm_bf16<4><<<blocks(waves * 32LL, BS), BS, 0, stream>>>(
            hbf, Wm0T, bm0, nullptr, mbf, N_NODES, H_DIM, M_DIM, 1, 0);
    }

    // --- logits: out = mbf @ Wm1 + bm1, store only first 40 cols ---
    {
        int waves = (N_NODES >> 4) * (L_PAD / 48);       // NT=3 -> 16x48 per wave
        k_gemm_bf16<3><<<blocks(waves * 32LL, BS), BS, 0, stream>>>(
            mbf, Wm1T, bm1, (float*)d_out, nullptr, N_NODES, M_DIM, L_PAD, 2, L_DIM);
    }
}
